// MultiRoundDistribution_44848048504926
// MI455X (gfx1250) — matrix operand order repaired and checked by reading, not compiled, not run on hardware
//
#include <hip/hip_runtime.h>
#include <math.h>

// MI455X / gfx1250, wave32. One fused kernel:
//   phase 1: WMMA projection S = X * [h0 | modes_h]  (f16 inputs, f32 acc)
//            + packed-category extraction (chains read exactly once)
//   phase 2: 1 thread per chain, incremental Metropolis on sufficient stats.

typedef __attribute__((ext_vector_type(16))) _Float16 v16h;
typedef __attribute__((ext_vector_type(8)))  float    v8f;

#define NCH  16384
#define LL   48
#define QQ   4
#define MM   4
#define FEAT (LL*QQ)   /* 192 */
#define TMAX 7

__device__ __forceinline__ unsigned int hash32(unsigned int x){
  x ^= x >> 16; x *= 0x7feb352du;
  x ^= x >> 15; x *= 0x846ca68bu;
  x ^= x >> 16;
  return x;
}

__device__ __forceinline__ float lse4(float a, float b, float c, float d){
  float mx = fmaxf(fmaxf(a,b), fmaxf(c,d));
  return mx + __logf(__expf(a-mx)+__expf(b-mx)+__expf(c-mx)+__expf(d-mx));
}

__device__ __forceinline__ float lse_masked(const float s[4], unsigned int mask){
  float mx = -3.0e38f;
  #pragma unroll
  for (int m = 0; m < MM; ++m)
    mx = fmaxf(mx, ((mask >> m) & 1u) ? s[m] : -3.0e38f);
  float sum = 0.f;
  #pragma unroll
  for (int m = 0; m < MM; ++m)
    sum += ((mask >> m) & 1u) ? __expf(s[m] - mx) : 0.f;
  return mx + __logf(sum);
}

// E = -( s0 + t*LSE_m(-s[m]) + sum_{r<t} LSE_{m in sel_r}(s[m]) )
__device__ __forceinline__ float energyS(float s0, const float s[4], int tval,
                                         const unsigned int* mask){
  if (tval < 0) return 0.f;               // t == -1: zero energy
  float acc = s0;
  if (tval > 0){
    float zneg = lse4(-s[0], -s[1], -s[2], -s[3]);
    acc += (float)tval * zneg;
    #pragma unroll
    for (int r = 0; r < TMAX; ++r)
      if (r < tval) acc += lse_masked(s, mask[r]);
  }
  return -acc;
}

__global__ __launch_bounds__(256)
void mrd_mcmc_kernel(const float* __restrict__ chains,
                     const float* __restrict__ h0,
                     const float* __restrict__ modes_h,
                     const int*   __restrict__ selected,
                     const int*   __restrict__ t_ptr,
                     const int*   __restrict__ nsw_ptr,
                     float* __restrict__ out)
{
  __shared__ float        Hc[(1+MM)*FEAT];   // col 0 = h0, col 1+m = modes_h[m]
  __shared__ float        Sld[256*8];        // per-chain scores (cols 0..4 used)
  __shared__ unsigned int Pk[256*3];         // packed categories, 2 bits/site

  const int tid       = threadIdx.x;
  const int lane      = tid & 31;
  const int wave      = tid >> 5;
  const int blockBase = blockIdx.x * 256;

  // ---- stage fields into LDS; zero the packed-category array -------------
  for (int i = tid; i < FEAT;    i += 256) Hc[i]        = h0[i];
  for (int i = tid; i < MM*FEAT; i += 256) Hc[FEAT + i] = modes_h[i];
  Pk[tid*3+0] = 0u; Pk[tid*3+1] = 0u; Pk[tid*3+2] = 0u;
  __syncthreads();

  // ---- build B operands: 6 K-chunks of the 192x16 field matrix ----------
  // B layout (16-bit, 32x16): lane holds column N = lane&15,
  // K = (lane>>4)*8 + elem (elem<8), 16 + (lane>>4)*8 + elem-8 (elem>=8) within each chunk of 32.
  const int colB = lane & 15;
  const int koff = (lane >> 4) << 3;        // 0 or 8
  v16h Bt[6];
  #pragma unroll
  for (int kc = 0; kc < 6; ++kc){
    v16h bb;
    #pragma unroll
    for (int j = 0; j < 16; ++j){
      int K = kc*32 + koff + (j < 8 ? j : j + 8);
      float v = (colB < 1+MM) ? Hc[colB*FEAT + K] : 0.0f;
      bb[j] = (_Float16)v;
    }
    Bt[kc] = bb;
  }

  // ---- phase 1: WMMA projection + category packing -----------------------
  // A layout (16-bit, 16x32): lane holds row M = lane&15,
  // K = khalf..khalf+7 (VGPR 0-3) and 16+khalf..16+khalf+7 (VGPR 4-7).
  const int rowA  = lane & 15;
  const int khalf = (lane >> 4) << 3;       // 0 or 8
  #pragma unroll
  for (int t16 = 0; t16 < 2; ++t16){
    const int chainLocal = wave*32 + t16*16 + rowA;
    const float* rowp = chains + (size_t)(blockBase + chainLocal) * FEAT;
    v8f acc = {};
    unsigned int pkLoc[3] = {0u, 0u, 0u};
    #pragma unroll
    for (int kc = 0; kc < 6; ++kc){
      float4 g0a = *(const float4*)(rowp + kc*32 + khalf);
      float4 g0b = *(const float4*)(rowp + kc*32 + khalf + 4);
      float4 g1a = *(const float4*)(rowp + kc*32 + 16 + khalf);
      float4 g1b = *(const float4*)(rowp + kc*32 + 16 + khalf + 4);

      // one-hot -> category (exact): c = y + 2z + 3w
      int l0 = (kc*32 + khalf) >> 2;
      int l1 = (kc*32 + 16 + khalf) >> 2;
      int c0 = (int)(g0a.y + 2.f*g0a.z + 3.f*g0a.w);
      int c1 = (int)(g0b.y + 2.f*g0b.z + 3.f*g0b.w);
      int c2 = (int)(g1a.y + 2.f*g1a.z + 3.f*g1a.w);
      int c3 = (int)(g1b.y + 2.f*g1b.z + 3.f*g1b.w);
      pkLoc[(l0  )>>4] |= (unsigned int)c0 << (((l0  )&15)*2);
      pkLoc[(l0+1)>>4] |= (unsigned int)c1 << (((l0+1)&15)*2);
      pkLoc[(l1  )>>4] |= (unsigned int)c2 << (((l1  )&15)*2);
      pkLoc[(l1+1)>>4] |= (unsigned int)c3 << (((l1+1)&15)*2);

      v16h aa;
      aa[ 0]=(_Float16)g0a.x; aa[ 1]=(_Float16)g0a.y; aa[ 2]=(_Float16)g0a.z; aa[ 3]=(_Float16)g0a.w;
      aa[ 4]=(_Float16)g0b.x; aa[ 5]=(_Float16)g0b.y; aa[ 6]=(_Float16)g0b.z; aa[ 7]=(_Float16)g0b.w;
      aa[ 8]=(_Float16)g1a.x; aa[ 9]=(_Float16)g1a.y; aa[10]=(_Float16)g1a.z; aa[11]=(_Float16)g1a.w;
      aa[12]=(_Float16)g1b.x; aa[13]=(_Float16)g1b.y; aa[14]=(_Float16)g1b.z; aa[15]=(_Float16)g1b.w;

      acc = __builtin_amdgcn_wmma_f32_16x16x32_f16(
              /*neg_a=*/false, aa, /*neg_b=*/false, Bt[kc],
              /*c_mod=*/(short)0, acc, /*reuse_a=*/false, /*reuse_b=*/false);
    }

    // C/D layout: VGPR v -> row M = v + (lane<16 ? 0 : 8), col N = lane&15
    const int mAdd = (lane < 16) ? 0 : 8;
    if (colB < 1+MM){
      #pragma unroll
      for (int v = 0; v < 8; ++v){
        int cl = wave*32 + t16*16 + v + mAdd;
        Sld[cl*8 + colB] = acc[v];
      }
    }
    atomicOr(&Pk[chainLocal*3+0], pkLoc[0]);
    atomicOr(&Pk[chainLocal*3+1], pkLoc[1]);
    atomicOr(&Pk[chainLocal*3+2], pkLoc[2]);
  }
  __syncthreads();

  // ---- phase 2: per-chain incremental Metropolis -------------------------
  const int chain = blockBase + tid;
  float s0 = Sld[tid*8 + 0];
  float s[4] = { Sld[tid*8+1], Sld[tid*8+2], Sld[tid*8+3], Sld[tid*8+4] };
  unsigned int pk[3] = { Pk[tid*3+0], Pk[tid*3+1], Pk[tid*3+2] };

  const int tval = t_ptr[0];
  const int nsw  = nsw_ptr[0];
  unsigned int mask[TMAX];
  #pragma unroll
  for (int r = 0; r < TMAX; ++r){
    unsigned int mk = 0u;
    if (r < tval){
      #pragma unroll
      for (int m = 0; m < MM; ++m)
        mk |= (selected[r*MM + m] != 0 ? 1u : 0u) << m;
    }
    mask[r] = mk;
  }

  const int nsteps = nsw * LL;
  float Ecur = energyS(s0, s, tval, mask);

  for (int step = 0; step < nsteps; ++step){
    // per-step key (uniform across all chains) -> site index (stays scalar)
    unsigned int hstep = hash32(0x9E3779B9u * (unsigned int)step + 0x6a09e667u);
    int idx = (int)(hstep % (unsigned int)LL);
    // per-(step,chain) -> proposal + acceptance uniform
    unsigned int hc = hash32(hstep ^ ((unsigned int)chain * 0x85ebca6bu) ^ 0xbb67ae85u);
    int cnew = (int)(hc & 3u);
    float u = (float)(hash32(hc ^ 0x3c6ef372u) >> 8) * (1.0f/16777216.0f);

    const int w  = idx >> 4;
    const int sh = (idx & 15) * 2;
    unsigned int pw = (w == 0) ? pk[0] : ((w == 1) ? pk[1] : pk[2]);
    int cold = (int)((pw >> sh) & 3u);

    const int base = idx * QQ;
    float sn0 = s0 + (Hc[base + cnew] - Hc[base + cold]);
    float sn[4];
    #pragma unroll
    for (int m = 0; m < MM; ++m)
      sn[m] = s[m] + (Hc[(1+m)*FEAT + base + cnew] - Hc[(1+m)*FEAT + base + cold]);

    float Enew = energyS(sn0, sn, tval, mask);
    float de = Enew - Ecur;                 // BETA = 1
    if (__expf(-de) > u){
      s0 = sn0;
      s[0] = sn[0]; s[1] = sn[1]; s[2] = sn[2]; s[3] = sn[3];
      Ecur = Enew;
      unsigned int nw = (pw & ~(3u << sh)) | ((unsigned int)cnew << sh);
      if (w == 0) pk[0] = nw; else if (w == 1) pk[1] = nw; else pk[2] = nw;
    }
  }

  // ---- write one-hot chains out ------------------------------------------
  float* orow = out + (size_t)chain * FEAT;
  #pragma unroll
  for (int l = 0; l < LL; ++l){
    int c = (int)((pk[l >> 4] >> ((l & 15) * 2)) & 3u);
    float4 v;
    v.x = (c == 0) ? 1.f : 0.f;
    v.y = (c == 1) ? 1.f : 0.f;
    v.z = (c == 2) ? 1.f : 0.f;
    v.w = (c == 3) ? 1.f : 0.f;
    *(float4*)(orow + l*4) = v;
  }
}

extern "C" void kernel_launch(void* const* d_in, const int* in_sizes, int n_in,
                              void* d_out, int out_size, void* d_ws, size_t ws_size,
                              hipStream_t stream){
  (void)in_sizes; (void)n_in; (void)out_size; (void)d_ws; (void)ws_size;
  const float* chains   = (const float*)d_in[0];
  const float* h0       = (const float*)d_in[1];
  const float* modes_h  = (const float*)d_in[2];
  const int*   selected = (const int*)  d_in[3];
  const int*   t_ptr    = (const int*)  d_in[4];
  const int*   nsw_ptr  = (const int*)  d_in[5];
  float* out = (float*)d_out;

  dim3 grid(NCH / 256), block(256);
  hipLaunchKernelGGL(mrd_mcmc_kernel, grid, block, 0, stream,
                     chains, h0, modes_h, selected, t_ptr, nsw_ptr, out);
}
